// GAT_9749575762831
// MI455X (gfx1250) — matrix operand order repaired and checked by reading, not compiled, not run on hardware
//
#include <hip/hip_runtime.h>

typedef __attribute__((ext_vector_type(16))) _Float16     v16h;
typedef __attribute__((ext_vector_type(8)))  float        v8f;
typedef __attribute__((ext_vector_type(4)))  float        v4f;
typedef __attribute__((ext_vector_type(4)))  int          v4i;
typedef __attribute__((ext_vector_type(8)))  int          v8i;
typedef __attribute__((ext_vector_type(4)))  unsigned int v4u;

#define B_    64
#define N_    1024
#define FIN_  128
#define FOUT_ 64
#define NEG_BIG_F (-9.0e15f)

// LDS row stride for staged adjacency: 128 data DWORDs + 4 pad DWORDs (TDM
// pad-on-load) -> 16B-aligned rows, bank-conflict-free b128 reads.
#define ADJ_ROW_  132
#define ADJ_TILE_ (16 * ADJ_ROW_)

// ---------------------------------------------------------------------------
// Kernel 1: Wh = atoms @ W  (f32 in, f16 out, f32 WMMA accumulate)
// One wave computes a 16x64 tile of Wh, stored TRANSPOSED WhT[(b*FOUT+f)*N+n].
// ---------------------------------------------------------------------------
__global__ __launch_bounds__(32) void gat_wh_kernel(
    const float* __restrict__ atoms, const float* __restrict__ W,
    _Float16* __restrict__ WhT) {
  const int tile = blockIdx.x;            // B * N/16 = 4096 tiles
  const int b    = tile >> 6;
  const int i0   = (tile & 63) << 4;
  const int lane = threadIdx.x;
  const int hi   = lane >> 4;
  const int lm   = lane & 15;

  v8f acc[4] = {v8f{}, v8f{}, v8f{}, v8f{}};

  for (int kt = 0; kt < 4; ++kt) {        // K = 128 in 4 steps of 32
    const float* arow = atoms + ((size_t)(b * N_ + i0 + lm)) * FIN_
                              + kt * 32 + (hi ? 8 : 0);
    v4f a0 = *(const v4f*)(arow + 0);
    v4f a1 = *(const v4f*)(arow + 4);
    v4f a2 = *(const v4f*)(arow + 16);
    v4f a3 = *(const v4f*)(arow + 20);
    v16h afrag;
#pragma unroll
    for (int t = 0; t < 4; ++t) {
      afrag[t]      = (_Float16)a0[t];
      afrag[4 + t]  = (_Float16)a1[t];
      afrag[8 + t]  = (_Float16)a2[t];
      afrag[12 + t] = (_Float16)a3[t];
    }
#pragma unroll
    for (int nt = 0; nt < 4; ++nt) {
      v16h bfrag;
#pragma unroll
      for (int h = 0; h < 16; ++h) {
        const int K = kt * 32 + (hi ? 8 : 0) + (h < 8 ? h : h + 8);
        bfrag[h] = (_Float16)W[K * FOUT_ + nt * 16 + lm];
      }
      acc[nt] = __builtin_amdgcn_wmma_f32_16x16x32_f16(
          false, afrag, false, bfrag, (short)0, acc[nt], false, false);
    }
  }

#pragma unroll
  for (int nt = 0; nt < 4; ++nt)
#pragma unroll
    for (int r = 0; r < 8; ++r) {
      const int n = i0 + r + (hi ? 8 : 0);
      const int f = nt * 16 + lm;
      WhT[((size_t)(b * FOUT_ + f)) * N_ + n] = (_Float16)acc[nt][r];
    }
}

// ---------------------------------------------------------------------------
// Kernel 2: s1[b,n] = Wh[b,n,:] . a[:64],  s2[b,n] = Wh[b,n,:] . a[64:128]
// ---------------------------------------------------------------------------
__global__ void gat_s_kernel(const _Float16* __restrict__ WhT,
                             const float* __restrict__ a,
                             float* __restrict__ s1, float* __restrict__ s2) {
  const int gid = blockIdx.x * blockDim.x + threadIdx.x;
  if (gid >= B_ * N_) return;
  const int b = gid >> 10, n = gid & (N_ - 1);
  const _Float16* base = WhT + (size_t)b * FOUT_ * N_ + n;
  float r1 = 0.f, r2 = 0.f;
#pragma unroll 8
  for (int f = 0; f < FOUT_; ++f) {
    const float w = (float)base[(size_t)f * N_];
    r1 += w * a[f];
    r2 += w * a[FOUT_ + f];
  }
  s1[gid] = r1;
  s2[gid] = r2;
}

// ---------------------------------------------------------------------------
// TDM: DMA a 16x128 int32 adjacency tile (row stride N_) into LDS, padding
// 4 DWORDs after every 128 DWORDs (row stride 132 in LDS).
// D# built per CDNA5 ISA; 6-arg builtin on this
// toolchain: (u32x4 g0, i32x8 g1, i32x4 g2, i32x4 g3, i32x8 pad, i32 cpol).
// ---------------------------------------------------------------------------
__device__ __forceinline__ void tdm_load_adj(const int* gptr,
                                             unsigned int lds_addr) {
  const unsigned long long ga = (unsigned long long)(size_t)gptr;
  v4u g0;
  g0[0] = 1u;                                  // count=1 (valid user D#)
  g0[1] = lds_addr;                            // bits 63:32  lds_addr
  g0[2] = (unsigned int)ga;                    // bits 95:64  global_addr lo
  g0[3] = (unsigned int)((ga >> 32) & 0x01FFFFFFu) | (2u << 30); // addr hi|type=2
  v8i g1;
  // data_size=2 (4B) | pad_enable | pad_interval=6 (128 dw) | pad_amount=3 (4 dw)
  g1[0] = (int)((2u << 16) | (1u << 20) | (6u << 22) | (3u << 25));
  g1[1] = (int)((N_ & 0xFFFFu) << 16);         // tensor_dim0[15:0]
  g1[2] = (int)((N_ >> 16) | ((N_ & 0xFFFFu) << 16)); // dim0 hi | dim1 lo
  g1[3] = (int)((N_ >> 16) | (128u << 16));    // dim1 hi | tile_dim0=128
  g1[4] = 16;                                  // tile_dim1=16, tile_dim2=0
  g1[5] = N_;                                  // tensor_dim0_stride lo
  g1[6] = 0;
  g1[7] = 0;
  const v4i z4 = {0, 0, 0, 0};                 // groups 2/3 unused (2-D tensor)
  const v8i z8 = {0, 0, 0, 0, 0, 0, 0, 0};
  __builtin_amdgcn_tensor_load_to_lds(g0, g1, z4, z4, z8, 0);
}

// ---------------------------------------------------------------------------
// Kernel 3: fused masked softmax + (attention @ Wh) + ELU, flash-style.
// Adjacency streamed by the Tensor Data Mover into double-buffered LDS;
// one DMA kept in flight while the wave does exp/softmax/WMMA work.
// ---------------------------------------------------------------------------
__global__ __launch_bounds__(32) void gat_attn_kernel(
    const int* __restrict__ adj, const _Float16* __restrict__ WhT,
    const float* __restrict__ s1, const float* __restrict__ s2,
    float* __restrict__ out) {
  __shared__ int adjlds[2][ADJ_TILE_];

  const int tile = blockIdx.x;            // 4096 tiles
  const int b    = tile >> 6;
  const int i0   = (tile & 63) << 4;
  const int lane = threadIdx.x;
  const int hi   = lane >> 4;
  const int lm   = lane & 15;
  const int off0 = hi ? 8 : 0;            // A-fragment K base for this half

  const float     s1row = s1[b * N_ + i0 + lm];
  const float*    s2b   = s2 + b * N_;
  const _Float16* whb   = WhT + (size_t)b * FOUT_ * N_;
  const int*      adjt  = adj + ((size_t)(b * N_ + i0)) * N_;  // tile row 0

  const unsigned int lds0 = (unsigned int)(size_t)&adjlds[0][0];
  const unsigned int lds1 = (unsigned int)(size_t)&adjlds[1][0];

  v8f acc[4] = {v8f{}, v8f{}, v8f{}, v8f{}};
  float m = NEG_BIG_F;
  float l = 0.f;

  tdm_load_adj(adjt, lds0);               // prime chunk 0

  for (int c = 0; c < 8; ++c) {           // 8 chunks of 128 columns
    if (c + 1 < 8) {                      // keep one DMA in flight
      tdm_load_adj(adjt + (c + 1) * 128, (c & 1) ? lds0 : lds1);
      __builtin_amdgcn_s_wait_tensorcnt(1);
    } else {
      __builtin_amdgcn_s_wait_tensorcnt(0);
    }
    const int* lrow = &adjlds[c & 1][lm * ADJ_ROW_];

#pragma unroll
    for (int jt = 0; jt < 4; ++jt) {      // 4 j-tiles of 32 per chunk
      const int j0  = c * 128 + jt * 32;  // global column base
      const int j0l = jt * 32;            // column base inside LDS tile

      if (jt + 1 < 4)                     // -> global_prefetch_b8 (WhT, L2-hot)
        __builtin_prefetch(whb + (size_t)lm * N_ + j0 + 32, 0, 0);

      // Adjacency from LDS (conflict-free b128s) + s2, A-fragment order.
      v4i ad[4]; v4f sv[4];
      ad[0] = *(const v4i*)(lrow + j0l + off0 + 0);
      ad[1] = *(const v4i*)(lrow + j0l + off0 + 4);
      ad[2] = *(const v4i*)(lrow + j0l + off0 + 16);
      ad[3] = *(const v4i*)(lrow + j0l + off0 + 20);
      sv[0] = *(const v4f*)(s2b + j0 + off0 + 0);
      sv[1] = *(const v4f*)(s2b + j0 + off0 + 4);
      sv[2] = *(const v4f*)(s2b + j0 + off0 + 16);
      sv[3] = *(const v4f*)(s2b + j0 + off0 + 20);

      float e[16];
      float mloc = NEG_BIG_F;
#pragma unroll
      for (int g = 0; g < 4; ++g)
#pragma unroll
        for (int t = 0; t < 4; ++t) {
          float ev = s1row + sv[g][t];
          ev = (ev > 0.f) ? ev : 0.1f * ev;        // leaky_relu(0.1)
          ev = (ad[g][t] > 0) ? ev : NEG_BIG_F;    // adjacency mask
          e[g * 4 + t] = ev;
          mloc = fmaxf(mloc, ev);
        }
      mloc = fmaxf(mloc, __shfl_xor(mloc, 16, 32));
      const float mnew  = fmaxf(m, mloc);
      const float scale = __expf(m - mnew);
      m = mnew;

      // Rescale accumulators (C layout: VGPR r holds row r + off0).
#pragma unroll
      for (int r = 0; r < 8; ++r) {
        const float fr = __shfl(scale, r + off0, 32);
#pragma unroll
        for (int nt = 0; nt < 4; ++nt) acc[nt][r] *= fr;
      }

      // P = exp(e - mnew), packed straight into the A fragment.
      v16h afrag;
      float psum = 0.f;
#pragma unroll
      for (int h = 0; h < 16; ++h) {
        const float ph = __expf(e[h] - mnew);      // masked -> underflows to 0
        psum += ph;
        afrag[h] = (_Float16)ph;
      }
      psum += __shfl_xor(psum, 16, 32);
      l = l * scale + psum;

      // B fragments from WhT (table K order per lane): two b128 loads each.
#pragma unroll
      for (int nt = 0; nt < 4; ++nt) {
        union { v16h v; v4i q[2]; } bf;
        const v4i* bp = (const v4i*)(whb + (size_t)(nt * 16 + lm) * N_
                                         + j0 + (hi ? 8 : 0));
        bf.q[0] = bp[0];
        bf.q[1] = bp[2];
        acc[nt] = __builtin_amdgcn_wmma_f32_16x16x32_f16(
            false, afrag, false, bf.v, (short)0, acc[nt], false, false);
      }
    }
  }

  // Normalize by l, apply ELU, store f32 output (B, N, FOUT) row-major.
#pragma unroll
  for (int r = 0; r < 8; ++r) {
    const float lrowv = __shfl(l, r + off0, 32);
    const float inv   = 1.f / lrowv;
    const int   n     = i0 + r + off0;
#pragma unroll
    for (int nt = 0; nt < 4; ++nt) {
      float v = acc[nt][r] * inv;
      v = (v > 0.f) ? v : (__expf(v) - 1.f);       // ELU(alpha=1)
      out[((size_t)(b * N_ + n)) * FOUT_ + nt * 16 + lm] = v;
    }
  }
}

// ---------------------------------------------------------------------------
// Launch. Inputs: d_in[0]=atoms f32, d_in[1]=adjacency i32, d_in[2]=W f32,
// d_in[3]=a f32. Workspace: WhT f16 (8 MB) | s1 (256 KB) | s2 (256 KB).
// ---------------------------------------------------------------------------
extern "C" void kernel_launch(void* const* d_in, const int* in_sizes, int n_in,
                              void* d_out, int out_size, void* d_ws, size_t ws_size,
                              hipStream_t stream) {
  const float* atoms = (const float*)d_in[0];
  const int*   adj   = (const int*)d_in[1];
  const float* W     = (const float*)d_in[2];
  const float* a     = (const float*)d_in[3];
  float*       out   = (float*)d_out;

  _Float16* WhT = (_Float16*)d_ws;
  float*    s1  = (float*)((char*)d_ws + (size_t)B_ * FOUT_ * N_ * sizeof(_Float16));
  float*    s2  = s1 + B_ * N_;

  const int tiles = B_ * (N_ / 16);       // 4096
  gat_wh_kernel<<<dim3(tiles), dim3(32), 0, stream>>>(atoms, W, WhT);
  gat_s_kernel<<<dim3((B_ * N_) / 256), dim3(256), 0, stream>>>(WhT, a, s1, s2);
  gat_attn_kernel<<<dim3(tiles), dim3(32), 0, stream>>>(adj, WhT, s1, s2, out);
}
